// RNN_747324310248
// MI455X (gfx1250) — matrix operand order repaired and checked by reading, not compiled, not run on hardware
//
#include <hip/hip_runtime.h>
#include <hip/hip_bf16.h>

typedef __attribute__((ext_vector_type(16))) _Float16 v16h;
typedef __attribute__((ext_vector_type(8)))  _Float16 v8h;
typedef __attribute__((ext_vector_type(8)))  float    v8f;

namespace {

constexpr int S = 512;
constexpr int B = 64;
constexpr int I = 256;
constexpr int H = 1024;
constexpr int WAVES_PER_BLK = 8;   // 256 threads, wave32
constexpr int NBLK_REC = 8;        // persistent blocks for the scan (64 waves)
constexpr int LDS_STRIDE = H + 8;  // f16 elems; 2064B row stride = 4 banks mod 64
constexpr size_t LDS_BYTES = (size_t)128 * LDS_STRIDE * sizeof(_Float16); // 258KB

union V16 { v16h v; v8h h[2]; };

// ---- f32 -> f16 converting loads (used only in the one-shot xproj GEMM) ----
__device__ __forceinline__ void cvt8(v16h& v, int base, const float4& a, const float4& b) {
  v[base + 0] = (_Float16)a.x; v[base + 1] = (_Float16)a.y;
  v[base + 2] = (_Float16)a.z; v[base + 3] = (_Float16)a.w;
  v[base + 4] = (_Float16)b.x; v[base + 5] = (_Float16)b.y;
  v[base + 6] = (_Float16)b.z; v[base + 7] = (_Float16)b.w;
}

// A-fragment (16x32 f16): lane L holds row = L%16,
// v[0..7] <- K = kA..kA+7 (kA = k0 + (L>>4)*8), v[8..15] <- K = kA+16..kA+23.
__device__ __forceinline__ v16h load_a_frag_f32(const float* __restrict__ rowPtr, int kA) {
  const float4* p0 = reinterpret_cast<const float4*>(rowPtr + kA);
  const float4* p1 = reinterpret_cast<const float4*>(rowPtr + kA + 16);
  float4 a0 = p0[0], a1 = p0[1];
  float4 b0 = p1[0], b1 = p1[1];
  v16h v;
  cvt8(v, 0, a0, a1);
  cvt8(v, 8, b0, b1);
  return v;
}

// ---- packed f16 fragment loads ----
// B-fragment (32x16 f16): lane L holds col = L%16, two 8-K chunks at p and p+16.
__device__ __forceinline__ v16h load_b_frag_f16(const _Float16* p) {
  V16 u;
  const v8h* q = reinterpret_cast<const v8h*>(p);
  u.h[0] = q[0];
  u.h[1] = q[2];
  return u.v;
}

// A-fragment from packed f16: two 16B chunks at kA and kA+16.
__device__ __forceinline__ v16h load_a_frag_f16(const _Float16* rowPtr, int kA) {
  V16 u;
  u.h[0] = *reinterpret_cast<const v8h*>(rowPtr + kA);
  u.h[1] = *reinterpret_cast<const v8h*>(rowPtr + kA + 16);
  return u.v;
}

// ---------------------------------------------------------------------------
// Weight pre-conversion: Wh (H*H) and Wx (H*I) f32 -> f16, once per call.
// ---------------------------------------------------------------------------
__global__ __launch_bounds__(256) void cvt_weights(
    const float* __restrict__ Wh, const float* __restrict__ Wx,
    _Float16* __restrict__ Wh16, _Float16* __restrict__ Wx16) {
  const int stride = gridDim.x * blockDim.x;
  const int i0 = blockIdx.x * blockDim.x + threadIdx.x;
  for (int i = i0; i < H * H; i += stride) Wh16[i] = (_Float16)Wh[i];
  for (int i = i0; i < H * I; i += stride) Wx16[i] = (_Float16)Wx[i];
}

// ---------------------------------------------------------------------------
// Kernel A: xproj[s*B+b, h] = sum_i x[s,b,i] * Wx16[h,i] + bx[h]
// One wave -> 16 rows x 64 cols strip; written into d_out's h_seq region.
// ---------------------------------------------------------------------------
__global__ __launch_bounds__(256) void xproj_wmma(
    const float* __restrict__ x, const _Float16* __restrict__ Wx16,
    const float* __restrict__ bx, float* __restrict__ out) {
  const int lane  = threadIdx.x & 31;
  const int r16   = lane & 15;
  const int halfK = lane >> 4;
  const int wave  = blockIdx.x * WAVES_PER_BLK + (threadIdx.x >> 5);
  const int nStrips = H / 64;            // 16
  const int mTile   = wave / nStrips;
  const int nStrip  = wave % nStrips;
  const int row0 = mTile * 16;
  const int col0 = nStrip * 64;

  const float* aRow = x + (size_t)(row0 + r16) * I;

  v8f acc[4] = {};
  for (int k0 = 0; k0 < I; k0 += 32) {
    v16h a = load_a_frag_f32(aRow, k0 + halfK * 8);
#pragma unroll
    for (int j = 0; j < 4; ++j) {
      const _Float16* bCol = Wx16 + (size_t)(col0 + j * 16 + r16) * I;
      v16h b = load_b_frag_f16(bCol + k0 + halfK * 8);
      acc[j] = __builtin_amdgcn_wmma_f32_16x16x32_f16(
          false, a, false, b, (short)0, acc[j], false, false);
    }
  }

#pragma unroll
  for (int j = 0; j < 4; ++j) {
    const int col = col0 + j * 16 + r16;
    const float bias = bx[col];
#pragma unroll
    for (int r = 0; r < 8; ++r) {
      const int row = row0 + r + 8 * halfK;
      out[(size_t)row * H + col] = acc[j][r] + bias;
    }
  }
}

// ---------------------------------------------------------------------------
// Grid barrier (sense-reversing, agent scope).
// ---------------------------------------------------------------------------
__global__ void ws_init(unsigned* ws) {
  if (threadIdx.x < 2) ws[threadIdx.x] = 0u;
}

__device__ __forceinline__ void grid_barrier(unsigned* cnt, unsigned* gen,
                                             unsigned nblocks) {
  __syncthreads();
  if (threadIdx.x == 0) {
    __threadfence();
    unsigned g = __hip_atomic_load(gen, __ATOMIC_RELAXED, __HIP_MEMORY_SCOPE_AGENT);
    unsigned arrived =
        __hip_atomic_fetch_add(cnt, 1u, __ATOMIC_ACQ_REL, __HIP_MEMORY_SCOPE_AGENT) + 1u;
    if (arrived == nblocks) {
      __hip_atomic_store(cnt, 0u, __ATOMIC_RELAXED, __HIP_MEMORY_SCOPE_AGENT);
      __hip_atomic_store(gen, g + 1u, __ATOMIC_RELEASE, __HIP_MEMORY_SCOPE_AGENT);
    } else {
      while (__hip_atomic_load(gen, __ATOMIC_ACQUIRE, __HIP_MEMORY_SCOPE_AGENT) == g) {
        __builtin_amdgcn_s_sleep(1);
      }
    }
  }
  __syncthreads();
}

// ---------------------------------------------------------------------------
// Kernel B: persistent scan. 8 blocks x 8 waves; wave w owns hidden columns
// [16w, 16w+16). Each block stages its 128 Wh columns (f16) into LDS once
// via CDNA5 async global->LDS DMA; h state is carried between steps as f16
// in a global ping-pong buffer (L2-resident).
// ---------------------------------------------------------------------------
__global__ __launch_bounds__(256) void rnn_scan_wmma(
    const _Float16* __restrict__ Wh16, const float* __restrict__ bh,
    float* __restrict__ out, _Float16* __restrict__ hbuf0,
    _Float16* __restrict__ hbuf1, unsigned* __restrict__ bar) {
  extern __shared__ _Float16 ldsW[];  // [128][LDS_STRIDE]

  const int tid   = threadIdx.x;
  const int lane  = tid & 31;
  const int r16   = lane & 15;
  const int halfK = lane >> 4;
  const int waveInBlk = tid >> 5;                            // 0..7
  const int wave  = blockIdx.x * WAVES_PER_BLK + waveInBlk;  // 0..63
  const int col0  = wave * 16;
  const int blkCol0 = blockIdx.x * 128;

  // One-time: stage this block's 128 columns of Wh16 into LDS with the
  // CDNA5 async DMA path (GLOBAL_LOAD_ASYNC_TO_LDS_B128, ASYNCcnt-tracked).
  {
    const int col  = tid >> 1;   // 0..127
    const int half = tid & 1;    // 0..1
    const _Float16* src = Wh16 + (size_t)(blkCol0 + col) * H + half * (H / 2);
    // LDS byte address of this thread's destination (dynamic-LDS base +
    // element offset); consistent with later ds_loads through ldsW.
    const unsigned dstBase = __builtin_amdgcn_groupstaticsize() +
        (unsigned)((col * LDS_STRIDE + half * (H / 2)) * sizeof(_Float16));
#pragma unroll
    for (int i = 0; i < (H / 2) / 8; ++i) {  // 64 async 16B transfers
      unsigned ldsAddr = dstBase + (unsigned)(i * 16);
      unsigned long long gaddr = (unsigned long long)(uintptr_t)(src + i * 8);
      asm volatile("global_load_async_to_lds_b128 %0, %1, off"
                   :: "v"(ldsAddr), "v"(gaddr) : "memory");
    }
    asm volatile("s_wait_asynccnt 0x0" ::: "memory");
  }
  __syncthreads();

  const _Float16* ldsCol = ldsW + (waveInBlk * 16 + r16) * LDS_STRIDE;
  const float biasv = bh[col0 + r16];
  float* const hlast = out + (size_t)S * B * H;

  for (int t = 0; t < S; ++t) {
    v8f acc[4] = {};
    if (t > 0) {  // h_0 == 0 (wave-uniform branch, EXEC stays all-ones)
      const _Float16* hprev = ((t - 1) & 1) ? hbuf1 : hbuf0;
      for (int k0 = 0; k0 < H; k0 += 32) {
        v16h bfrag = load_b_frag_f16(ldsCol + k0 + halfK * 8);
#pragma unroll
        for (int m = 0; m < 4; ++m) {
          v16h afrag = load_a_frag_f16(hprev + (size_t)(m * 16 + r16) * H,
                                       k0 + halfK * 8);
          acc[m] = __builtin_amdgcn_wmma_f32_16x16x32_f16(
              false, afrag, false, bfrag, (short)0, acc[m], false, false);
        }
      }
    }

    float* cur = out + (size_t)t * (B * H);
    _Float16* hcur = (t & 1) ? hbuf1 : hbuf0;
    const int col = col0 + r16;
#pragma unroll
    for (int m = 0; m < 4; ++m) {
#pragma unroll
      for (int r = 0; r < 8; ++r) {
        const int row = m * 16 + r + 8 * halfK;
        const size_t idx = (size_t)row * H + col;
        float v = tanhf(acc[m][r] + biasv + cur[idx]);  // fuse xproj + bias
        cur[idx]  = v;              // f32 h_seq output
        hcur[idx] = (_Float16)v;    // f16 state for next step
        if (t == S - 1) hlast[idx] = v;
      }
    }

    grid_barrier(&bar[0], &bar[1], (unsigned)gridDim.x);
  }
}

}  // namespace

extern "C" void kernel_launch(void* const* d_in, const int* in_sizes, int n_in,
                              void* d_out, int out_size, void* d_ws, size_t ws_size,
                              hipStream_t stream) {
  (void)in_sizes; (void)n_in; (void)out_size; (void)ws_size;
  const float* x  = (const float*)d_in[0];
  const float* Wx = (const float*)d_in[1];
  const float* bx = (const float*)d_in[2];
  const float* Wh = (const float*)d_in[3];
  const float* bh = (const float*)d_in[4];
  float* out = (float*)d_out;

  // Workspace layout (all 256B-aligned): barrier | Wh16 | Wx16 | hbuf0 | hbuf1
  uintptr_t base = (uintptr_t)d_ws;
  unsigned*  bar   = (unsigned*)base;
  _Float16*  Wh16  = (_Float16*)(base + 256);
  _Float16*  Wx16  = Wh16 + (size_t)H * H;
  _Float16*  hbuf0 = Wx16 + (size_t)H * I;
  _Float16*  hbuf1 = hbuf0 + (size_t)B * H;

  ws_init<<<1, 64, 0, stream>>>(bar);
  cvt_weights<<<512, 256, 0, stream>>>(Wh, Wx, Wh16, Wx16);

  // Phase 1: xproj GEMM. (S*B/16) M-tiles x (H/64) N-strips waves.
  const int wavesA = (S * B / 16) * (H / 64);  // 32768 waves
  xproj_wmma<<<wavesA / WAVES_PER_BLK, 256, 0, stream>>>(x, Wx16, bx, out);

  // Phase 2: persistent recurrent scan with Wh staged into LDS via async DMA.
  rnn_scan_wmma<<<NBLK_REC, 256, LDS_BYTES, stream>>>(Wh16, bh, out, hbuf0,
                                                      hbuf1, bar);
}
